// TopologicalSignature_75273596829931
// MI455X (gfx1250) — matrix operand order repaired and checked by reading, not compiled, not run on hardware
//
#include <hip/hip_runtime.h>
#include <stdint.h>

// ---------------------------------------------------------------------------
// Problem constants (reference shapes are fixed: x is [2048, 4096] fp32).
// ---------------------------------------------------------------------------
#define N_PTS   2048
#define DIM     4096
#define N_OUT   (N_PTS - 1)   // 2047 MST edge weights, ascending

// WMMA types (gfx1250, wave32)
typedef __attribute__((ext_vector_type(16))) __bf16 v16bf;
typedef __attribute__((ext_vector_type(8)))  float  v8f;

// ---------------------------------------------------------------------------
// bf16 helpers (split-precision: x = hi + lo, both representable in bf16)
// ---------------------------------------------------------------------------
__device__ __forceinline__ unsigned short f32_to_bf16_rn(float f) {
    unsigned int u = __float_as_uint(f);
    unsigned int r = u + 0x7FFFu + ((u >> 16) & 1u);   // round-to-nearest-even
    return (unsigned short)(r >> 16);
}
__device__ __forceinline__ float bf16_to_f32(unsigned short h) {
    return __uint_as_float(((unsigned int)h) << 16);
}

// ---------------------------------------------------------------------------
// Kernel 1: per-row squared norm + split fp32 row into (hi, lo) bf16 arrays.
// One block per row, 256 threads, each thread covers 16 strided columns.
// ---------------------------------------------------------------------------
__global__ __launch_bounds__(256)
void split_sqnorm_kernel(const float* __restrict__ x,
                         unsigned short* __restrict__ xhi,
                         unsigned short* __restrict__ xlo,
                         float* __restrict__ sq) {
    const int row = blockIdx.x;
    const int t   = threadIdx.x;
    const size_t base = (size_t)row * DIM;

    float acc = 0.0f;
    for (int c = t; c < DIM; c += 256) {
        float v = x[base + c];
        unsigned short hi = f32_to_bf16_rn(v);
        float fhi = bf16_to_f32(hi);
        unsigned short lo = f32_to_bf16_rn(v - fhi);
        xhi[base + c] = hi;
        xlo[base + c] = lo;
        acc += v * v;
    }
    __shared__ float red[256];
    red[t] = acc;
    __syncthreads();
    for (int s = 128; s > 0; s >>= 1) {
        if (t < s) red[t] += red[t + s];
        __syncthreads();
    }
    if (t == 0) sq[row] = red[0];
}

// ---------------------------------------------------------------------------
// WMMA tile loaders. X is row-major [N_PTS][DIM] bf16 (as ushort).
//
// A-matrix 16x32 bf16 layout (ISA 7.12.2): lane m (0..15) holds row m.
//   lanes 0-15 : VGPR0-3 = K 0..7,  VGPR4-7 = K 16..23
//   lanes 16-31: VGPR0-3 = K 8..15, VGPR4-7 = K 24..31
// -> two contiguous 16-byte chunks at k0+8*half and k0+16+8*half.
//
// B-matrix 32x16 bf16 layout: lane n holds column n (= row J+n of X, since
// B = X^T tile). Same K order as A: lanes 0-15 hold K 0..7,16..23,
// lanes 16-31 hold K 8..15,24..31 -> chunks at k0+8*half and k0+16+8*half.
// ---------------------------------------------------------------------------
__device__ __forceinline__ v16bf load_tileA(const unsigned short* __restrict__ X,
                                            int row0, int k0, int lane) {
    const int m    = lane & 15;
    const int half = lane >> 4;
    const unsigned short* p = X + (size_t)(row0 + m) * DIM + k0 + 8 * half;
    union { uint4 u[2]; v16bf v; } U;
    U.u[0] = *(const uint4*)(p);        // K chunk 0 (8 bf16)
    U.u[1] = *(const uint4*)(p + 16);   // K chunk 1 (8 bf16)
    return U.v;
}
__device__ __forceinline__ v16bf load_tileB(const unsigned short* __restrict__ X,
                                            int col0, int k0, int lane) {
    const int n    = lane & 15;
    const int half = lane >> 4;
    const unsigned short* p = X + (size_t)(col0 + n) * DIM + k0 + 8 * half;
    union { uint4 u[2]; v16bf v; } U;
    U.u[0] = *(const uint4*)(p);
    U.u[1] = *(const uint4*)(p + 16);
    return U.v;
}

__device__ __forceinline__ v8f wmma_bf16(v16bf a, v16bf b, v8f c) {
    // (neg_a, A, neg_b, B, c_mod, C, reuse_a, reuse_b)
    return __builtin_amdgcn_wmma_f32_16x16x32_bf16(false, a, false, b,
                                                   (short)0, c, false, false);
}

// ---------------------------------------------------------------------------
// Kernel 2: distance matrix via split-bf16 WMMA GEMM.
// Grid (16, 32), block 256 (8 waves). Block tile = 128 rows x 64 cols.
// Wave tile = 32x32 (2x2 accumulators). dot = hi*hi + hi*lo + lo*hi.
// D[i][j] = sqrt(max(sq[i] + sq[j] - 2*dot, 0)).
// ---------------------------------------------------------------------------
__global__ __launch_bounds__(256)
void gram_dist_kernel(const unsigned short* __restrict__ xhi,
                      const unsigned short* __restrict__ xlo,
                      const float* __restrict__ sq,
                      float* __restrict__ D) {
    const int lane = threadIdx.x & 31;
    const int wv   = threadIdx.x >> 5;      // 0..7
    const int wr   = wv >> 1;               // 0..3  (4 wave-rows)
    const int wc   = wv & 1;                // 0..1  (2 wave-cols)

    const int I = blockIdx.x * 128 + wr * 32;   // output row origin (32 rows)
    const int J = blockIdx.y * 64  + wc * 32;   // output col origin (32 cols)

    v8f acc[2][2] = {};   // [ti][tj] 16x16 f32 tiles

    for (int k0 = 0; k0 < DIM; k0 += 32) {
        v16bf ah[2], al[2], bh[2], bl[2];
        ah[0] = load_tileA(xhi, I,      k0, lane);
        ah[1] = load_tileA(xhi, I + 16, k0, lane);
        al[0] = load_tileA(xlo, I,      k0, lane);
        al[1] = load_tileA(xlo, I + 16, k0, lane);
        bh[0] = load_tileB(xhi, J,      k0, lane);
        bh[1] = load_tileB(xhi, J + 16, k0, lane);
        bl[0] = load_tileB(xlo, J,      k0, lane);
        bl[1] = load_tileB(xlo, J + 16, k0, lane);

#pragma unroll
        for (int ti = 0; ti < 2; ++ti)
#pragma unroll
            for (int tj = 0; tj < 2; ++tj) {
                acc[ti][tj] = wmma_bf16(ah[ti], bh[tj], acc[ti][tj]); // hi*hi
                acc[ti][tj] = wmma_bf16(ah[ti], bl[tj], acc[ti][tj]); // hi*lo
                acc[ti][tj] = wmma_bf16(al[ti], bh[tj], acc[ti][tj]); // lo*hi
            }
    }

    // Epilogue. C/D 16x16 f32 layout: lanes 0-15 -> M = vgpr, N = lane;
    // lanes 16-31 -> M = 8 + vgpr, N = lane - 16.
    const int nloc  = lane & 15;
    const int mbase = (lane >> 4) * 8;
#pragma unroll
    for (int ti = 0; ti < 2; ++ti) {
#pragma unroll
        for (int tj = 0; tj < 2; ++tj) {
            const int col = J + tj * 16 + nloc;
            const float sqc = sq[col];
#pragma unroll
            for (int v = 0; v < 8; ++v) {
                const int row = I + ti * 16 + mbase + v;
                float d2 = sq[row] + sqc - 2.0f * acc[ti][tj][v];
                D[(size_t)row * N_PTS + col] = __builtin_sqrtf(fmaxf(d2, 0.0f));
            }
        }
    }
}

// ---------------------------------------------------------------------------
// Kernel 3: Prim's MST on the dense distance matrix. One block, 1024 threads.
// minW[] lives in LDS; wave32 shfl_xor reductions over packed (weight,idx).
// Emits the N-1 MST edge weights (unsorted) into wts[].
// ---------------------------------------------------------------------------
__global__ __launch_bounds__(1024)
void prim_kernel(const float* __restrict__ D, float* __restrict__ wts) {
    __shared__ float minW[N_PTS];
    __shared__ unsigned char inMST[N_PTS];
    __shared__ unsigned long long red[32];
    __shared__ unsigned long long bestShared;

    const int t    = threadIdx.x;
    const int lane = t & 31;
    const int wv   = t >> 5;

    for (int j = t; j < N_PTS; j += 1024) {
        minW[j]  = D[j];          // row 0 = start vertex
        inMST[j] = 0;
    }
    if (t == 0) inMST[0] = 1;
    __syncthreads();

    for (int it = 0; it < N_OUT; ++it) {
        // pack (float bits << 32 | index); distances >= 0 so uint order works
        unsigned long long best = ~0ull;
        for (int j = t; j < N_PTS; j += 1024) {
            float w = inMST[j] ? __builtin_inff() : minW[j];
            unsigned long long key =
                ((unsigned long long)__float_as_uint(w) << 32) | (unsigned)j;
            best = key < best ? key : best;
        }
#pragma unroll
        for (int d = 16; d > 0; d >>= 1) {
            unsigned long long o = __shfl_xor(best, d, 32);
            best = o < best ? o : best;
        }
        if (lane == 0) red[wv] = best;
        __syncthreads();
        if (wv == 0) {
            unsigned long long b = red[lane];
#pragma unroll
            for (int d = 16; d > 0; d >>= 1) {
                unsigned long long o = __shfl_xor(b, d, 32);
                b = o < b ? o : b;
            }
            if (lane == 0) bestShared = b;
        }
        __syncthreads();

        const unsigned long long b = bestShared;
        const int v = (int)(b & 0xFFFFFFFFu);
        if (t == 0) {
            wts[it]  = __uint_as_float((unsigned)(b >> 32));
            inMST[v] = 1;
        }
        // relax with the new vertex's distance row (L2-resident)
        const float* __restrict__ row = D + (size_t)v * N_PTS;
        for (int j = t; j < N_PTS; j += 1024) {
            float d = row[j];
            if (d < minW[j]) minW[j] = d;
        }
        __syncthreads();
    }
}

// ---------------------------------------------------------------------------
// Kernel 4: bitonic sort of the 2047 MST weights (pad to 2048 with +inf).
// ---------------------------------------------------------------------------
__global__ __launch_bounds__(1024)
void sort_kernel(const float* __restrict__ wts, float* __restrict__ out) {
    __shared__ float s[N_PTS];
    const int t = threadIdx.x;
    for (int i = t; i < N_PTS; i += 1024)
        s[i] = (i < N_OUT) ? wts[i] : __builtin_inff();
    __syncthreads();

    for (int k = 2; k <= N_PTS; k <<= 1) {
        for (int j = k >> 1; j > 0; j >>= 1) {
            for (int idx = t; idx < N_PTS; idx += 1024) {
                int l = idx ^ j;
                if (l > idx) {
                    bool asc = ((idx & k) == 0);
                    float a = s[idx], b2 = s[l];
                    if ((a > b2) == asc) { s[idx] = b2; s[l] = a; }
                }
            }
            __syncthreads();
        }
    }
    for (int i = t; i < N_OUT; i += 1024) out[i] = s[i];
}

// ---------------------------------------------------------------------------
// Launch: x(fp32) -> split bf16 -> WMMA distance matrix -> Prim -> sort.
// Workspace layout (bytes):
//   sq   : N_PTS * 4                     =   8 KB
//   xhi  : N_PTS * DIM * 2               =  16 MB
//   xlo  : N_PTS * DIM * 2               =  16 MB
//   D    : N_PTS * N_PTS * 4             =  16 MB
//   wts  : N_PTS * 4                     =   8 KB
// ---------------------------------------------------------------------------
extern "C" void kernel_launch(void* const* d_in, const int* in_sizes, int n_in,
                              void* d_out, int out_size, void* d_ws, size_t ws_size,
                              hipStream_t stream) {
    (void)in_sizes; (void)n_in; (void)out_size; (void)ws_size;

    const float* x = (const float*)d_in[0];

    char* ws = (char*)d_ws;
    float*          sq  = (float*)ws;                 ws += (size_t)N_PTS * sizeof(float) * 2; // keep 16B align slack
    unsigned short* xhi = (unsigned short*)ws;        ws += (size_t)N_PTS * DIM * sizeof(unsigned short);
    unsigned short* xlo = (unsigned short*)ws;        ws += (size_t)N_PTS * DIM * sizeof(unsigned short);
    float*          D   = (float*)ws;                 ws += (size_t)N_PTS * N_PTS * sizeof(float);
    float*          wts = (float*)ws;

    split_sqnorm_kernel<<<N_PTS, 256, 0, stream>>>(x, xhi, xlo, sq);
    gram_dist_kernel<<<dim3(N_PTS / 128, N_PTS / 64), 256, 0, stream>>>(xhi, xlo, sq, D);
    prim_kernel<<<1, 1024, 0, stream>>>(D, wts);
    sort_kernel<<<1, 1024, 0, stream>>>(wts, (float*)d_out);
}
